// ModulatedConv3d_6786048327702
// MI455X (gfx1250) — matrix operand order repaired and checked by reading, not compiled, not run on hardware
//
#include <hip/hip_runtime.h>

typedef __attribute__((ext_vector_type(16))) __bf16 v16bf;
typedef __attribute__((ext_vector_type(8)))  float  v8f;
typedef __attribute__((ext_vector_type(4)))  unsigned v4u;
typedef __attribute__((ext_vector_type(8)))  unsigned v8u;

#define NS    4
#define INC   128
#define OUTC  128
#define DDIM  16
#define HDIM  64
#define WDIM  64
#define LDIM  512
#define KTAPS 27
#define KTOT  (KTAPS * INC)        // 3456
#define NCHUNK (KTOT / 32)         // 108
#define PLANE (DDIM * HDIM * WDIM) // 65536 elems per channel
#define HW    (HDIM * WDIM)        // 4096

__device__ __forceinline__ unsigned short f2bf(float f) {
    unsigned u = __float_as_uint(f);
    unsigned r = u + 0x7FFFu + ((u >> 16) & 1u);   // round-to-nearest-even
    return (unsigned short)(r >> 16);
}

// two f32 -> packed bf16x2 : bias then one v_perm_b32 grabs both high halves
__device__ __forceinline__ unsigned pack2bf(float a, float b) {
    unsigned ua = __float_as_uint(a) + 0x8000u;
    unsigned ub = __float_as_uint(b) + 0x8000u;
    return __builtin_amdgcn_perm(ub, ua, 0x07060302u); // {ub[31:16], ua[31:16]}
}

// ---------------------------------------------------------------- style
__global__ __launch_bounds__(256) void style_kernel(
    const float* __restrict__ latent, const float* __restrict__ fc_w,
    const float* __restrict__ fc_b, float* __restrict__ style)
{
    int idx = blockIdx.x * 256 + threadIdx.x;   // 0..511
    if (idx >= NS * INC) return;
    int n = idx >> 7, c = idx & 127;
    const float* lp = latent + (size_t)n * LDIM;
    const float* wp = fc_w + (size_t)c * LDIM;
    float sum = fc_b[c];
    #pragma unroll 4
    for (int l = 0; l < LDIM; ++l) sum += lp[l] * wp[l];
    style[idx] = sum;
}

// ---------------------------------------------------- modulate + demod
// wmod[n][o][tap*128 + i]  (bf16), GEMM-friendly K layout (tap-major)
__global__ __launch_bounds__(128) void wmod_kernel(
    const float* __restrict__ weight, const float* __restrict__ style,
    unsigned short* __restrict__ wmod)
{
    __shared__ float red[128];
    const int bo = blockIdx.x;          // n*128 + o
    const int n = bo >> 7, o = bo & 127;
    const int i = threadIdx.x;          // channel

    const float s = style[n * INC + i];
    const float* wp = weight + ((size_t)o * INC + i) * KTAPS;

    float wv[KTAPS];
    float sum = 0.f;
    #pragma unroll
    for (int t = 0; t < KTAPS; ++t) { wv[t] = wp[t] * s; sum += wv[t] * wv[t]; }

    red[i] = sum;
    __syncthreads();
    #pragma unroll
    for (int st = 64; st > 0; st >>= 1) {
        if (i < st) red[i] += red[i + st];
        __syncthreads();
    }
    const float demod = rsqrtf(red[0] + 1e-8f);

    unsigned short* op = wmod + (size_t)bo * KTOT + i;
    #pragma unroll
    for (int t = 0; t < KTAPS; ++t) op[t * INC] = f2bf(wv[t] * demod);
}

// --------------------------------------------------------- implicit GEMM
// Block: one (n,d) and two consecutive h rows -> N-tile = 128 positions,
// all 128 outC. 8 waves, each 16(M) x 128(N) = 8 WMMA tiles, K = 3456.
// A tile (128x32 bf16) is DMA'd per chunk by the Tensor Data Mover.
__global__ __launch_bounds__(256) void conv_kernel(
    const float* __restrict__ x, const unsigned short* __restrict__ wmod,
    float* __restrict__ out)
{
    __shared__ __align__(16) unsigned short As[2][128 * 32]; // [m][k]  8KB/buf
    __shared__ __align__(16) unsigned short Bs[2][128 * 32]; // [p][k]  8KB/buf

    const int t   = threadIdx.x;
    const int blk = blockIdx.x;
    const int n    = blk >> 9;          // 512 row-pairs per sample
    const int pair = blk & 511;
    const int d    = pair >> 5;
    const int h0   = (pair & 31) * 2;   // rows h0, h0+1

    const float*          xn  = x    + (size_t)n * INC * PLANE;
    const unsigned short* wn  = wmod + (size_t)n * OUTC * KTOT;
    const unsigned long long wn64 = (unsigned long long)(uintptr_t)wn;

    // B staging decode: thread owns one position p and 16 channels
    const int p   = t & 127;            // p = rh*64 + w
    const int rh  = p >> 6;
    const int w   = p & 63;
    const int cg  = t >> 7;             // 0/1 -> channels cg*16 .. +15

    // fragment decode (wave32 WMMA layouts)
    const int lane = t & 31;
    const int wid  = t >> 5;            // wave 0..7
    const int m0   = wid * 16;
    const int arow = m0 + (lane & 15);
    const int akb  = (lane >> 4) * 8;   // A: K base 0/8 (upper half +16)
    const int bkb  = (lane >> 4) * 8;   // B: K base 0/8 (upper half +16)
    const int bcol = lane & 15;

    v8f acc[8] = {};

    // TDM: one descriptor-driven DMA moves the whole 128x32 bf16 A tile.
    // D# per CDNA5 ISA (2D, groups 0+1, VADDR2/3 NULL).
    auto issueTDM = [&](int buf, int cc) {
        const unsigned long long ga = wn64 + (unsigned long long)cc * 64ull;
        v4u g0;
        g0[0] = 1u;                                   // count=1, user mode
        g0[1] = (unsigned)(uintptr_t)&As[buf][0];     // lds_addr (bytes)
        g0[2] = (unsigned)ga;                         // global_addr[31:0]
        g0[3] = (unsigned)(ga >> 32) | (2u << 30);    // addr[56:32] | type=2
        v8u g1;
        g1[0] = 1u << 16;                             // data_size = 2 bytes
        g1[1] = ((unsigned)KTOT & 0xFFFFu) << 16;     // tensor_dim0 lo16
        g1[2] = ((unsigned)KTOT >> 16) | (128u << 16);// dim0 hi | tensor_dim1=128
        g1[3] = 32u << 16;                            // dim1 hi(0) | tile_dim0=32
        g1[4] = 128u;                                 // tile_dim1=128, tile_dim2=0
        g1[5] = (unsigned)KTOT;                       // tensor_dim0_stride lo32
        g1[6] = 0u;                                   // stride hi | dim1_stride
        g1[7] = 0u;
        asm volatile("tensor_load_to_lds %0, %1" :: "s"(g0), "s"(g1) : "memory");
    };

    // branchless uniform chunk decode: chunk cc -> tap, channel base
    auto gloadB = [&](int cc, float* f) {
        const int tap = cc >> 2;
        const int i0  = (cc & 3) * 32;
        const int kd  = (tap * 57) >> 9;      // /9
        const int r9  = tap - kd * 9;
        const int kh  = (r9 * 171) >> 9;      // /3
        const int kw  = r9 - kh * 3;
        const int dz = d + kd - 1, hz = h0 + rh + kh - 1, wz = w + kw - 1;
        const bool ok = (dz >= 0) && (dz < DDIM) && (hz >= 0) && (hz < HDIM) &&
                        (wz >= 0) && (wz < WDIM);
        const float* px = xn + (size_t)(i0 + cg * 16) * PLANE + dz * HW + hz * WDIM + wz;
        #pragma unroll
        for (int cc2 = 0; cc2 < 16; ++cc2)
            f[cc2] = ok ? px[(size_t)cc2 * PLANE] : 0.0f;
    };
    auto storeB = [&](int buf, const float* f) {
        uint4 pk0, pk1;
        pk0.x = pack2bf(f[0],  f[1]);  pk0.y = pack2bf(f[2],  f[3]);
        pk0.z = pack2bf(f[4],  f[5]);  pk0.w = pack2bf(f[6],  f[7]);
        pk1.x = pack2bf(f[8],  f[9]);  pk1.y = pack2bf(f[10], f[11]);
        pk1.z = pack2bf(f[12], f[13]); pk1.w = pack2bf(f[14], f[15]);
        *(uint4*)(void*)&Bs[buf][p * 32 + cg * 16]     = pk0;
        *(uint4*)(void*)&Bs[buf][p * 32 + cg * 16 + 8] = pk1;
    };

    // ---- prologue: chunk 0
    {
        float f[16];
        if (wid == 0) issueTDM(0, 0);
        gloadB(0, f);
        storeB(0, f);
        if (wid == 0) __builtin_amdgcn_s_wait_tensorcnt(0);
    }
    __syncthreads();

    // ---- main loop
    for (int c = 0; c < NCHUNK; ++c) {
        const int buf = c & 1;
        const bool more = (c + 1 < NCHUNK);
        float nB[16];
        if (more) {
            if (wid == 0) issueTDM(buf ^ 1, c + 1);   // TDM prefetch of A
            gloadB(c + 1, nB);
        }

        union Frag { uint4 u[2]; v16bf v; };
        Frag af;
        af.u[0] = *(const uint4*)(const void*)&As[buf][arow * 32 + akb];
        af.u[1] = *(const uint4*)(const void*)&As[buf][arow * 32 + akb + 16];
        #pragma unroll
        for (int nt = 0; nt < 8; ++nt) {
            Frag bfr;
            const int pc = nt * 16 + bcol;
            bfr.u[0] = *(const uint4*)(const void*)&Bs[buf][pc * 32 + bkb];
            bfr.u[1] = *(const uint4*)(const void*)&Bs[buf][pc * 32 + bkb + 16];
            acc[nt] = __builtin_amdgcn_wmma_f32_16x16x32_bf16(
                false, af.v, false, bfr.v, (short)0, acc[nt], false, false);
        }

        if (more) storeB(buf ^ 1, nB);
        if (wid == 0) __builtin_amdgcn_s_wait_tensorcnt(0);
        __syncthreads();
    }

    // ---- epilogue: C tile — VGPR r: lanes 0-15 -> M=r, lanes 16-31 -> M=8+r
    float* on = out + (size_t)n * OUTC * PLANE + (size_t)d * HW + h0 * WDIM;
    const int rofs = (lane >> 4) * 8;
    #pragma unroll
    for (int nt = 0; nt < 8; ++nt) {
        const int col = nt * 16 + bcol;    // col = rh*64 + w
        #pragma unroll
        for (int r = 0; r < 8; ++r) {
            const int m = m0 + rofs + r;
            on[(size_t)m * PLANE + col] = acc[nt][r];
        }
    }
}

// ------------------------------------------------------------- launcher
extern "C" void kernel_launch(void* const* d_in, const int* in_sizes, int n_in,
                              void* d_out, int out_size, void* d_ws, size_t ws_size,
                              hipStream_t stream) {
    const float* x      = (const float*)d_in[0];
    const float* latent = (const float*)d_in[1];
    const float* weight = (const float*)d_in[2];
    const float* fc_w   = (const float*)d_in[3];
    const float* fc_b   = (const float*)d_in[4];
    float* out = (float*)d_out;

    float*          style = (float*)d_ws;                            // 2 KB
    unsigned short* wmodp = (unsigned short*)((char*)d_ws + 4096);   // ~3.4 MB

    style_kernel<<<2, 256, 0, stream>>>(latent, fc_w, fc_b, style);
    wmod_kernel<<<NS * OUTC, 128, 0, stream>>>(weight, style, wmodp);
    conv_kernel<<<NS * DDIM * (HDIM / 2), 256, 0, stream>>>(x, wmodp, out);
}
